// TransformerDecoderLayer_43336220016862
// MI455X (gfx1250) — matrix operand order repaired and checked by reading, not compiled, not run on hardware
//
#include <hip/hip_runtime.h>
#include <hip/hip_bf16.h>
#include <stdint.h>

// ---------------------------------------------------------------------------
// MSDeformAttn transformer decoder layer for gfx1250 (MI455X), wave32 + WMMA.
// All dense GEMMs -> v_wmma_f32_16x16x32_f16 (f16 in, f32 accumulate).
// Value-proj GEMM runs single-pass over its 178 MB streaming A operand
// (NT=16 -> one block covers all 256 output columns).
// ---------------------------------------------------------------------------

typedef __attribute__((ext_vector_type(16))) _Float16 v16h;
typedef __attribute__((ext_vector_type(8)))  _Float16 v8h;
typedef __attribute__((ext_vector_type(8)))  float    v8f;

#define NB      8
#define LQ      2048
#define LIN     21760
#define DMODEL  256
#define DFFN    1024
#define NHEAD   8
#define DHEAD   32

// -------------------------- weight transpose (f32[K,N] -> f16[N,K]) --------
__global__ void wt_f16_kernel(const float* __restrict__ W,
                              _Float16* __restrict__ Wt, int K, int N) {
  int t = blockIdx.x * blockDim.x + threadIdx.x;
  if (t >= K * N) return;
  int n = t / K, k = t - n * K;
  Wt[t] = (_Float16)W[k * N + n];
}

// -------------------------- WMMA GEMM --------------------------------------
// C[M,N] = A[M,K] * B[K,N] + bias,  B given transposed as Bt[N,K] (f16).
// AMODE 0: A16 f16 row-major.  AMODE 1: A = f16(A0 + A1) fused from two f32.
// EPI bits: 1 = store f32 C32, 2 = store f16 C16, 4 = ReLU.
// NT: number of 16-wide column tiles per wave (wave strip = 16 x 16*NT).
// Block = 8 waves stacked in M -> block tile 128 x 16*NT.
template <int AMODE, int EPI, int NT>
__global__ __launch_bounds__(256) void gemm_wmma_kernel(
    const _Float16* __restrict__ A16,
    const float* __restrict__ A0, const float* __restrict__ A1,
    const _Float16* __restrict__ Bt, const float* __restrict__ bias,
    float* __restrict__ C32, _Float16* __restrict__ C16,
    int M, int N, int K) {
  const int lane = threadIdx.x & 31;
  const int wid  = threadIdx.x >> 5;
  const int row0 = blockIdx.y * 128 + wid * 16;
  const int col0 = blockIdx.x * (NT * 16);
  if (row0 >= M) return;

  const int half = lane >> 4;   // 0: lanes 0-15, 1: lanes 16-31
  const int r    = lane & 15;

  v8f acc[NT] = {};

  for (int k0 = 0; k0 < K; k0 += 32) {
    // ---- A fragment: 16x32, lane<16: row r, K = k0+{0..7} & k0+{16..23};
    //                          lane>=16: row r, K = k0+{8..15} & k0+{24..31}.
    v16h a;
    if (AMODE == 0) {
      const _Float16* pa = A16 + (size_t)(row0 + r) * K + k0 + half * 8;
      v8h lo = *(const v8h*)(pa);
      v8h hi = *(const v8h*)(pa + 16);
#pragma unroll
      for (int j = 0; j < 8; ++j) { a[j] = lo[j]; a[8 + j] = hi[j]; }
    } else {
      const float* p0 = A0 + (size_t)(row0 + r) * K + k0 + half * 8;
      const float* p1 = A1 + (size_t)(row0 + r) * K + k0 + half * 8;
      v8f lo0 = *(const v8f*)(p0), hi0 = *(const v8f*)(p0 + 16);
      v8f lo1 = *(const v8f*)(p1), hi1 = *(const v8f*)(p1 + 16);
#pragma unroll
      for (int j = 0; j < 8; ++j) {
        a[j]     = (_Float16)(lo0[j] + lo1[j]);
        a[8 + j] = (_Float16)(hi0[j] + hi1[j]);
      }
    }

    // ---- B fragments: 32x16; lane<16: col r, K = k0+{0..7} & k0+{16..23};
    //                          lane>=16: col r, K = k0+{8..15} & k0+{24..31}.
    const _Float16* pb = Bt + (size_t)(col0 + r) * K + k0 + half * 8;
#pragma unroll
    for (int t = 0; t < NT; ++t) {
      v8h blo = *(const v8h*)(pb + (size_t)t * 16 * K);
      v8h bhi = *(const v8h*)(pb + (size_t)t * 16 * K + 16);
      v16h b = __builtin_shufflevector(blo, bhi, 0, 1, 2, 3, 4, 5, 6, 7, 8, 9, 10, 11, 12, 13, 14, 15);
      acc[t] = __builtin_amdgcn_wmma_f32_16x16x32_f16(false, a, false, b,
                                                      (short)0, acc[t], false,
                                                      false);
    }
  }

  // ---- epilogue: C layout = VGPR i -> row (i + 8*half), col = lane&15.
#pragma unroll
  for (int t = 0; t < NT; ++t) {
    const int c  = col0 + t * 16 + r;
    const float bv = bias[c];
#pragma unroll
    for (int i = 0; i < 8; ++i) {
      const int row = row0 + i + half * 8;
      float val = acc[t][i] + bv;
      if (EPI & 4) val = fmaxf(val, 0.0f);
      const size_t idx = (size_t)row * N + c;
      if (EPI & 1) C32[idx] = val;
      if (EPI & 2) C16[idx] = (_Float16)val;
    }
  }
}

// -------------------------- softmax over 16 points (per n,q,head) ----------
__global__ void softmax16_kernel(float* __restrict__ attn, int rows) {
  int t = blockIdx.x * blockDim.x + threadIdx.x;
  if (t >= rows) return;
  float* p = attn + (size_t)t * 16;
  float v[16], m = -3.4e38f;
#pragma unroll
  for (int i = 0; i < 16; ++i) { v[i] = p[i]; m = fmaxf(m, v[i]); }
  float s = 0.f;
#pragma unroll
  for (int i = 0; i < 16; ++i) { v[i] = __expf(v[i] - m); s += v[i]; }
  const float inv = 1.0f / s;
#pragma unroll
  for (int i = 0; i < 16; ++i) p[i] = v[i] * inv;
}

// -------------------------- deformable bilinear sampling -------------------
// One wave per (n, head, q); lane = channel (DHEAD == 32 == wave32).
// v16: value tensor f16 laid out [n][lin][head][DHEAD]  (== GEMM output rows).
__global__ __launch_bounds__(256) void msdeform_sample_kernel(
    const _Float16* __restrict__ v16, const float* __restrict__ off,
    const float* __restrict__ attnw, const float* __restrict__ refp,
    _Float16* __restrict__ samp16) {
  const int gw   = (int)((blockIdx.x * blockDim.x + threadIdx.x) >> 5);
  const int lane = threadIdx.x & 31;
  const int q = gw & (LQ - 1);
  const int h = (gw >> 11) & (NHEAD - 1);
  const int n = gw >> 14;

  const int HL[4] = {128, 64, 32, 16};
  const int WL[4] = {128, 64, 32, 16};
  const int S0[4] = {0, 16384, 20480, 21504};

  const size_t qrow = (size_t)n * LQ + q;
  const float* offp = off + qrow * 256 + h * 32;        // h stride = L*P*2 = 32
  const float* ap   = attnw + (qrow * NHEAD + h) * 16;  // 16 softmax weights
  const float* rp   = refp + qrow * 8;                  // [L][2]

  float acc = 0.f;
#pragma unroll
  for (int l = 0; l < 4; ++l) {
    const int Wl = WL[l], Hl = HL[l];
    const float rx = rp[l * 2 + 0], ry = rp[l * 2 + 1];
    const _Float16* vl =
        v16 + ((size_t)n * LIN + S0[l]) * DMODEL + h * DHEAD + lane;
#pragma unroll
    for (int p = 0; p < 4; ++p) {
      const float aw = ap[l * 4 + p];
      const float ox = offp[(l * 4 + p) * 2 + 0];
      const float oy = offp[(l * 4 + p) * 2 + 1];
      // loc = ref + off/(W,H);  x = loc_x*W - 0.5 = ref_x*W + ox - 0.5
      const float x = rx * (float)Wl + ox - 0.5f;
      const float y = ry * (float)Hl + oy - 0.5f;
      const float xf = floorf(x), yf = floorf(y);
      const float wx = x - xf, wy = y - yf;
      const int x0 = (int)xf, y0 = (int)yf;
      const float w00 = (1.f - wx) * (1.f - wy) * aw;
      const float w01 = wx * (1.f - wy) * aw;
      const float w10 = (1.f - wx) * wy * aw;
      const float w11 = wx * wy * aw;
      const bool xin0 = (x0 >= 0) & (x0 < Wl);
      const bool xin1 = (x0 + 1 >= 0) & (x0 + 1 < Wl);
      if (y0 >= 0 && y0 < Hl) {
        const size_t rbase = (size_t)(y0 * Wl) * DMODEL;
        if (xin0) acc += w00 * (float)vl[rbase + (size_t)x0 * DMODEL];
        if (xin1) acc += w01 * (float)vl[rbase + (size_t)(x0 + 1) * DMODEL];
      }
      if (y0 + 1 >= 0 && y0 + 1 < Hl) {
        const size_t rbase = (size_t)((y0 + 1) * Wl) * DMODEL;
        if (xin0) acc += w10 * (float)vl[rbase + (size_t)x0 * DMODEL];
        if (xin1) acc += w11 * (float)vl[rbase + (size_t)(x0 + 1) * DMODEL];
      }
    }
  }
  samp16[qrow * DMODEL + h * DHEAD + lane] = (_Float16)acc;
}

// -------------------------- residual add + LayerNorm (one wave per row) ----
__global__ __launch_bounds__(256) void add_ln_kernel(
    const float* __restrict__ a, const float* __restrict__ b,
    const float* __restrict__ g, const float* __restrict__ be,
    float* __restrict__ out32, _Float16* __restrict__ out16, int rows) {
  const int w    = (int)((blockIdx.x * blockDim.x + threadIdx.x) >> 5);
  const int lane = threadIdx.x & 31;
  if (w >= rows) return;
  const float* pa = a + (size_t)w * DMODEL;
  const float* pb = b + (size_t)w * DMODEL;
  float v[8];
  float s = 0.f;
#pragma unroll
  for (int i = 0; i < 8; ++i) {
    v[i] = pa[lane + i * 32] + pb[lane + i * 32];
    s += v[i];
  }
#pragma unroll
  for (int o = 16; o > 0; o >>= 1) s += __shfl_xor(s, o, 32);
  const float mu = s * (1.0f / 256.0f);
  float var = 0.f;
#pragma unroll
  for (int i = 0; i < 8; ++i) { const float d = v[i] - mu; var += d * d; }
#pragma unroll
  for (int o = 16; o > 0; o >>= 1) var += __shfl_xor(var, o, 32);
  const float rs = rsqrtf(var * (1.0f / 256.0f) + 1e-5f);
#pragma unroll
  for (int i = 0; i < 8; ++i) {
    const int c = lane + i * 32;
    const float y = (v[i] - mu) * rs * g[c] + be[c];
    if (out32) out32[(size_t)w * DMODEL + c] = y;
    if (out16) out16[(size_t)w * DMODEL + c] = (_Float16)y;
  }
}

// ---------------------------------------------------------------------------
extern "C" void kernel_launch(void* const* d_in, const int* in_sizes, int n_in,
                              void* d_out, int out_size, void* d_ws,
                              size_t ws_size, hipStream_t stream) {
  (void)in_sizes; (void)n_in; (void)out_size; (void)ws_size;

  const float* src   = (const float*)d_in[0];
  const float* pos   = (const float*)d_in[1];
  const float* mem   = (const float*)d_in[2];
  const float* posm  = (const float*)d_in[3];
  const float* refp  = (const float*)d_in[4];
  // d_in[5], d_in[6]: static geometry (baked in as constants)
  const float* W_off = (const float*)d_in[7];
  const float* b_off = (const float*)d_in[8];
  const float* W_att = (const float*)d_in[9];
  const float* b_att = (const float*)d_in[10];
  const float* W_val = (const float*)d_in[11];
  const float* b_val = (const float*)d_in[12];
  const float* W_out = (const float*)d_in[13];
  const float* b_out = (const float*)d_in[14];
  const float* g1    = (const float*)d_in[15];
  const float* be1   = (const float*)d_in[16];
  const float* W_ff1 = (const float*)d_in[17];
  const float* b_ff1 = (const float*)d_in[18];
  const float* W_ff2 = (const float*)d_in[19];
  const float* b_ff2 = (const float*)d_in[20];
  const float* g2    = (const float*)d_in[21];
  const float* be2   = (const float*)d_in[22];

  // ---- workspace bump allocator (256B aligned) ----
  uint8_t* ws = (uint8_t*)d_ws;
  size_t cur = 0;
  auto alloc = [&](size_t bytes) -> void* {
    cur = (cur + 255) & ~(size_t)255;
    void* p = ws + cur;
    cur += bytes;
    return p;
  };

  const int MQ = NB * LQ;    // 16384
  const int MV = NB * LIN;   // 174080

  _Float16* Wv_t   = (_Float16*)alloc((size_t)DMODEL * DMODEL * 2);
  _Float16* Woff_t = (_Float16*)alloc((size_t)DMODEL * DMODEL * 2);
  _Float16* Watt_t = (_Float16*)alloc((size_t)DMODEL * 128 * 2);
  _Float16* Wout_t = (_Float16*)alloc((size_t)DMODEL * DMODEL * 2);
  _Float16* Wf1_t  = (_Float16*)alloc((size_t)DMODEL * DFFN * 2);
  _Float16* Wf2_t  = (_Float16*)alloc((size_t)DFFN * DMODEL * 2);
  _Float16* v16    = (_Float16*)alloc((size_t)MV * DMODEL * 2);
  float*    off32  = (float*)alloc((size_t)MQ * DMODEL * 4);
  float*    attn32 = (float*)alloc((size_t)MQ * 128 * 4);
  _Float16* samp16 = (_Float16*)alloc((size_t)MQ * DMODEL * 2);
  float*    src2   = (float*)alloc((size_t)MQ * DMODEL * 4);
  float*    x32    = (float*)alloc((size_t)MQ * DMODEL * 4);
  _Float16* x16    = (_Float16*)alloc((size_t)MQ * DMODEL * 2);
  _Float16* fh16   = (_Float16*)alloc((size_t)MQ * DFFN * 2);
  float*    h2     = (float*)alloc((size_t)MQ * DMODEL * 4);

  // ---- 1. transpose + f16-ize all weights ----
  auto wt = [&](const float* W, _Float16* Wt, int K, int N) {
    int tot = K * N;
    wt_f16_kernel<<<(tot + 255) / 256, 256, 0, stream>>>(W, Wt, K, N);
  };
  wt(W_val, Wv_t, DMODEL, DMODEL);
  wt(W_off, Woff_t, DMODEL, DMODEL);
  wt(W_att, Watt_t, DMODEL, 128);
  wt(W_out, Wout_t, DMODEL, DMODEL);
  wt(W_ff1, Wf1_t, DMODEL, DFFN);
  wt(W_ff2, Wf2_t, DFFN, DMODEL);

  // ---- 2. value proj: v = f16((mem+posm) @ W_val + b_val) ----
  // NT=16 -> single pass over the 178 MB streaming f32 A operand.
  gemm_wmma_kernel<1, 2, 16><<<dim3(1, MV / 128), 256, 0, stream>>>(
      nullptr, mem, posm, Wv_t, b_val, nullptr, v16, MV, DMODEL, DMODEL);

  // ---- 3. offsets: off = (src+pos) @ W_off + b_off ----
  gemm_wmma_kernel<1, 1, 4><<<dim3(DMODEL / 64, MQ / 128), 256, 0, stream>>>(
      nullptr, src, pos, Woff_t, b_off, off32, nullptr, MQ, DMODEL, DMODEL);

  // ---- 4. attention logits: (src+pos) @ W_attn + b_attn ----
  gemm_wmma_kernel<1, 1, 4><<<dim3(128 / 64, MQ / 128), 256, 0, stream>>>(
      nullptr, src, pos, Watt_t, b_att, attn32, nullptr, MQ, 128, DMODEL);

  // ---- 5. softmax over L*P = 16 per (n,q,head) ----
  {
    int rows = MQ * NHEAD;  // 131072
    softmax16_kernel<<<(rows + 255) / 256, 256, 0, stream>>>(attn32, rows);
  }

  // ---- 6. deformable bilinear sampling -> samp16 [n,q,(h,dh)] ----
  {
    int waves = NB * NHEAD * LQ;  // 131072
    msdeform_sample_kernel<<<waves / 8, 256, 0, stream>>>(v16, off32, attn32,
                                                          refp, samp16);
  }

  // ---- 7. output proj: src2 = samp @ W_out + b_out ----
  gemm_wmma_kernel<0, 1, 4><<<dim3(DMODEL / 64, MQ / 128), 256, 0, stream>>>(
      samp16, nullptr, nullptr, Wout_t, b_out, src2, nullptr, MQ, DMODEL,
      DMODEL);

  // ---- 8. x = LN(src + src2) -> x32 (residual) + x16 (FFN A operand) ----
  add_ln_kernel<<<(MQ * 32 + 255) / 256, 256, 0, stream>>>(src, src2, g1, be1,
                                                           x32, x16, MQ);

  // ---- 9. FFN1: fh = relu(x @ W_ff1 + b_ff1) stored f16 ----
  gemm_wmma_kernel<0, 6, 4><<<dim3(DFFN / 64, MQ / 128), 256, 0, stream>>>(
      x16, nullptr, nullptr, Wf1_t, b_ff1, nullptr, fh16, MQ, DFFN, DMODEL);

  // ---- 10. FFN2: h2 = fh @ W_ff2 + b_ff2 ----
  gemm_wmma_kernel<0, 1, 4><<<dim3(DMODEL / 64, MQ / 128), 256, 0, stream>>>(
      fh16, nullptr, nullptr, Wf2_t, b_ff2, h2, nullptr, MQ, DMODEL, DFFN);

  // ---- 11. out = LN(x + h2) ----
  add_ln_kernel<<<(MQ * 32 + 255) / 256, 256, 0, stream>>>(
      x32, h2, g2, be2, (float*)d_out, nullptr, MQ);
}
